// MultiHeadAttention_31490700214530
// MI455X (gfx1250) — matrix operand order repaired and checked by reading, not compiled, not run on hardware
//
#include <hip/hip_runtime.h>

// ---------------------------------------------------------------------------
// MultiHeadAttention (B=4, N=1024, D=1024, H=16, hd=64) for MI455X / gfx1250.
// All matmul FLOPs on v_wmma_f32_16x16x32_bf16 (fp32 accum), wave32, 4-wave
// blocks. bf16 tile staging uses GLOBAL_LOAD_ASYNC_TO_LDS_B128 (ASYNCcnt).
// Pipeline:
//   k1: Y = X @ W^T for {V,Q,K}; epilogue: 2D axial RoPE + RMSNorm on Q,K.
//       Q,K -> bf16 [B,H,N,64]; V -> bf16 [B,H,64,N] (d-major for PV B-frags).
//   k2: flash attention w/ soft-cap tanh + online softmax, WMMA for S and PV.
//       out -> bf16 [B,N,D].
//   k3: OUT = AO @ Wo^T, fp32 store to d_out.
// ---------------------------------------------------------------------------

#define DIM   1024
#define NHEAD 16
#define HD    64
#define NTOK  1024
#define BATCH 4

typedef __bf16 bf16;
typedef __attribute__((ext_vector_type(16))) __bf16 v16bf;
typedef __attribute__((ext_vector_type(8)))  __bf16 v8bf;
typedef __attribute__((ext_vector_type(8)))  float  v8f;

__device__ __forceinline__ v8f zero8() {
  v8f z = {0.f, 0.f, 0.f, 0.f, 0.f, 0.f, 0.f, 0.f};
  return z;
}

// Async 16B global -> LDS copy (CDNA5 Tensor/Async path, tracked by ASYNCcnt).
// LDS address = low 32 bits of the generic shared pointer (flat-aperture rule).
__device__ __forceinline__ void async_copy_b128(void* lds_dst, const void* gsrc) {
  unsigned ldsaddr = (unsigned)(unsigned long long)lds_dst;
  asm volatile("global_load_async_to_lds_b128 %0, %1, off"
               :
               : "v"(ldsaddr), "v"(gsrc)
               : "memory");
}

__device__ __forceinline__ void wait_async0() {
#if __has_builtin(__builtin_amdgcn_s_wait_asynccnt)
  __builtin_amdgcn_s_wait_asynccnt(0);
#else
  asm volatile("s_wait_asynccnt 0" ::: "memory");
#endif
}

// Build a v16bf fragment from two contiguous 16-byte LDS chunks.
// A-frag layout (ISA 7.12.2, 16-bit A 16x32): lane half h, elem e -> K = e + 8*((e>>3)+h)
//   => elems 0..7  contiguous at K = 8*h,  elems 8..15 contiguous at K = 16+8*h.
// B-frag layout: same K order as A: lane half h, elem e -> K = e + 8*((e>>3)+h)
//   => elems 0..7  contiguous at K = 8*h,  elems 8..15 contiguous at K = 16+8*h.
__device__ __forceinline__ v16bf pack2(const bf16* p0, const bf16* p1) {
  union { v16bf v; v8bf h[2]; } u;
  u.h[0] = *(const v8bf*)p0;
  u.h[1] = *(const v8bf*)p1;
  return u.v;
}

__device__ __forceinline__ v8f wmma_bf16(v16bf a, v16bf b, v8f c) {
  // (neg_a, A, neg_b, B, c_mod, C, reuse_a, reuse_b)
  return __builtin_amdgcn_wmma_f32_16x16x32_bf16(false, a, false, b,
                                                 (short)0, c, false, false);
}

__device__ __forceinline__ void cvt_store8(bf16* dst, float4 a, float4 b) {
  v8bf c;
  c[0] = (bf16)a.x; c[1] = (bf16)a.y; c[2] = (bf16)a.z; c[3] = (bf16)a.w;
  c[4] = (bf16)b.x; c[5] = (bf16)b.y; c[6] = (bf16)b.z; c[7] = (bf16)b.w;
  *(v8bf*)dst = c;
}

// ---------------------------------------------------------------------------
// Kernel 1: Y = X @ W^T (4096 x 1024 x 1024), col-tile == one head (64 dims).
// grid (BN/64=64, H=16, 3 mats {V,Q,K}), block 128 (4 waves, 16 rows/wave).
// fp32 inputs must be converted to bf16, so staging stays on the
// load->cvt->ds_store path here.
// ---------------------------------------------------------------------------
__global__ __launch_bounds__(128)
void qkv_rope_norm_kernel(const float* __restrict__ x,
                          const float* __restrict__ Wq,
                          const float* __restrict__ Wk,
                          const float* __restrict__ Wv,
                          const float* __restrict__ qnw,
                          const float* __restrict__ knw,
                          const int*   __restrict__ rope,
                          bf16* __restrict__ qb,
                          bf16* __restrict__ kb,
                          bf16* __restrict__ vb) {
  __shared__ __align__(16) bf16 lds_x[64 * 40];   // 64 rows x 32 k (pad->40)
  __shared__ __align__(16) bf16 lds_w[64 * 40];   // 64 e-rows x 32 k

  const int tid  = threadIdx.x;
  const int lane = tid & 31, wave = tid >> 5;
  const int m_   = lane & 15, half = lane >> 4;
  const int rowBase = blockIdx.x * 64;
  const int h    = blockIdx.y;
  const int eBase = h * 64;
  const int mat  = blockIdx.z;                    // 0=V, 1=Q, 2=K
  const float* W = (mat == 0) ? Wv : (mat == 1) ? Wq : Wk;

  v8f acc[4] = {zero8(), zero8(), zero8(), zero8()};

  const int chunk = tid & 3;       // which 8-float column chunk
  const int rhalf = tid >> 2;      // 0..31

  for (int kc = 0; kc < DIM; kc += 32) {
#pragma unroll
    for (int p = 0; p < 2; ++p) {
      const int row = p * 32 + rhalf;
      const float4* xs = (const float4*)(x + (size_t)(rowBase + row) * DIM + kc + chunk * 8);
      cvt_store8(&lds_x[row * 40 + chunk * 8], xs[0], xs[1]);
      const float4* ws = (const float4*)(W + (size_t)(eBase + row) * DIM + kc + chunk * 8);
      cvt_store8(&lds_w[row * 40 + chunk * 8], ws[0], ws[1]);
    }
    __syncthreads();

    const bf16* arow = &lds_x[(wave * 16 + m_) * 40];
    const v16bf afrag = pack2(arow + 8 * half, arow + 16 + 8 * half);
#pragma unroll
    for (int t = 0; t < 4; ++t) {
      const bf16* brow = &lds_w[(t * 16 + m_) * 40];
      const v16bf bfrag = pack2(brow + 8 * half, brow + 16 + 8 * half);
      acc[t] = wmma_bf16(afrag, bfrag, acc[t]);
    }
    __syncthreads();
  }

  // Epilogue. C/D layout: slot i, half -> row (i + 8*half); col = t*16 + m_.
  const float* nw = (mat == 1) ? qnw : knw;
#pragma unroll
  for (int i = 0; i < 8; ++i) {
    const int gr = rowBase + wave * 16 + i + 8 * half;  // flat row in [0,4096)
    const int bb = gr >> 10;
    const int n  = gr & (NTOK - 1);

    if (mat == 0) {
      // V: store d-major [B,H,64,N]
#pragma unroll
      for (int t = 0; t < 4; ++t) {
        const int d = t * 16 + m_;
        vb[((size_t)(bb * NHEAD + h) * HD + d) * NTOK + n] = (bf16)acc[t][i];
      }
    } else {
      const int ridx = rope[n];
      const float pos_y = (float)(ridx >> 5);
      const float pos_x = (float)(ridx & 31);
      float rv[4];
#pragma unroll
      for (int t = 0; t < 4; ++t) {
        const int d = t * 16 + m_;
        const int j = (d & 31) >> 1;
        // inv_freq = 10000^(-j/16) = exp2(-j * log2(10000)/16)
        const float freq = exp2f(-(float)j * 0.83048258f);
        const float pos  = (d < 32) ? pos_y : pos_x;
        const float ang  = pos * freq;
        const float c = __cosf(ang), s = __sinf(ang);
        const float v0 = acc[t][i];
        const float pr = __shfl_xor(v0, 1, 32);   // pair partner (d^1), same half
        const float rot = (d & 1) ? (v0 * c + pr * s) : (v0 * c - pr * s);
        rv[t] = (ridx >= 0) ? rot : v0;
      }
      // RMSNorm over the 64 head dims (4 in-lane vals x 16 lanes in half)
      float ss = rv[0] * rv[0] + rv[1] * rv[1] + rv[2] * rv[2] + rv[3] * rv[3];
#pragma unroll
      for (int msk = 1; msk <= 8; msk <<= 1) ss += __shfl_xor(ss, msk, 32);
      const float rms = sqrtf(ss * (1.0f / 64.0f) + 1e-6f);
      const float inv = 1.0f / rms;
      bf16* dst = (mat == 1) ? qb : kb;
#pragma unroll
      for (int t = 0; t < 4; ++t) {
        const int d = t * 16 + m_;
        dst[((size_t)(bb * NHEAD + h) * NTOK + n) * HD + d] =
            (bf16)(rv[t] * inv * nw[d]);
      }
    }
  }
}

// ---------------------------------------------------------------------------
// Kernel 2: flash attention with soft-cap tanh. grid (N/64=16, H, B), block 128.
// Q/K/V tiles are bf16 already -> staged with async global->LDS b128 copies.
// ---------------------------------------------------------------------------
__global__ __launch_bounds__(128)
void flash_attn_kernel(const bf16* __restrict__ qb,
                       const bf16* __restrict__ kb,
                       const bf16* __restrict__ vb,
                       bf16* __restrict__ ao) {
  __shared__ __align__(16) bf16 lds_q [64 * 72];
  __shared__ __align__(16) bf16 lds_k [64 * 72];
  __shared__ __align__(16) bf16 lds_vt[64 * 72];  // [d][key]
  __shared__ __align__(16) bf16 lds_p [64 * 72];

  const int tid  = threadIdx.x;
  const int lane = tid & 31, wave = tid >> 5;
  const int m_   = lane & 15, half = lane >> 4;
  const int qbase = blockIdx.x * 64;
  const int h = blockIdx.y, b = blockIdx.z;
  const size_t headQK = (size_t)(b * NHEAD + h) * NTOK * HD;
  const size_t headV  = (size_t)(b * NHEAD + h) * HD * NTOK;

  // Load Q tile (64 x 64 bf16) via async copies
#pragma unroll
  for (int c = 0; c < 4; ++c) {
    const int id = tid + 128 * c;
    const int row = id >> 3, cc = id & 7;
    async_copy_b128(&lds_q[row * 72 + cc * 8],
                    qb + headQK + (size_t)(qbase + row) * HD + cc * 8);
  }
  wait_async0();
  __syncthreads();

  v16bf aq[2];
  {
    const bf16* arow = &lds_q[(wave * 16 + m_) * 72];
#pragma unroll
    for (int s = 0; s < 2; ++s)
      aq[s] = pack2(arow + 32 * s + 8 * half, arow + 32 * s + 16 + 8 * half);
  }

  v8f oacc[4] = {zero8(), zero8(), zero8(), zero8()};
  float mrun[8], lrun[8];
#pragma unroll
  for (int i = 0; i < 8; ++i) { mrun[i] = -1e30f; lrun[i] = 0.f; }

  const float inv_cap_scale = 0.125f / 50.0f;   // (1/sqrt(64)) / SOFT_CAP

  for (int kt = 0; kt < NTOK / 64; ++kt) {
    const int kbase = kt * 64;
    __syncthreads();
#pragma unroll
    for (int c = 0; c < 4; ++c) {
      const int id = tid + 128 * c;
      const int row = id >> 3, cc = id & 7;
      async_copy_b128(&lds_k[row * 72 + cc * 8],
                      kb + headQK + (size_t)(kbase + row) * HD + cc * 8);
      async_copy_b128(&lds_vt[row * 72 + cc * 8],
                      vb + headV + (size_t)row * NTOK + kbase + cc * 8);
    }
    if (kt + 1 < NTOK / 64) {   // global_prefetch_b8 of next K/V tiles
      __builtin_prefetch(kb + headQK + (size_t)(kbase + 64) * HD, 0, 3);
      __builtin_prefetch(vb + headV + kbase + 64, 0, 3);
    }
    wait_async0();
    __syncthreads();

    // S = Q K^T  (keys = B-matrix columns; K-dim = head dim, 2 x 32 steps)
    v8f st[4] = {zero8(), zero8(), zero8(), zero8()};
#pragma unroll
    for (int t = 0; t < 4; ++t) {
#pragma unroll
      for (int s = 0; s < 2; ++s) {
        const bf16* brow = &lds_k[(t * 16 + m_) * 72 + 32 * s];
        const v16bf bfrag = pack2(brow + 8 * half, brow + 16 + 8 * half);
        st[t] = wmma_bf16(aq[s], bfrag, st[t]);
      }
    }

    // soft-cap + online softmax; write P (bf16) to LDS for C->A re-layout
#pragma unroll
    for (int i = 0; i < 8; ++i) {
      float sv[4], mx = -1e30f;
#pragma unroll
      for (int t = 0; t < 4; ++t) {
        sv[t] = 50.0f * tanhf(st[t][i] * inv_cap_scale);
        mx = fmaxf(mx, sv[t]);
      }
#pragma unroll
      for (int msk = 1; msk <= 8; msk <<= 1)
        mx = fmaxf(mx, __shfl_xor(mx, msk, 32));
      const float mnew  = fmaxf(mrun[i], mx);
      const float alpha = __expf(mrun[i] - mnew);
      float ls = 0.f;
      const int prow = (wave * 16 + i + 8 * half) * 72;
#pragma unroll
      for (int t = 0; t < 4; ++t) {
        const float p = __expf(sv[t] - mnew);
        ls += p;
        lds_p[prow + t * 16 + m_] = (bf16)p;
      }
#pragma unroll
      for (int msk = 1; msk <= 8; msk <<= 1)
        ls += __shfl_xor(ls, msk, 32);
      lrun[i] = lrun[i] * alpha + ls;
      mrun[i] = mnew;
#pragma unroll
      for (int t = 0; t < 4; ++t) oacc[t][i] *= alpha;
    }
    __syncthreads();

    // O += P V   (K-dim = keys, 2 x 32 steps; B from d-major V in LDS)
#pragma unroll
    for (int s = 0; s < 2; ++s) {
      const bf16* arow = &lds_p[(wave * 16 + m_) * 72 + 32 * s];
      const v16bf ap = pack2(arow + 8 * half, arow + 16 + 8 * half);
#pragma unroll
      for (int t = 0; t < 4; ++t) {
        const bf16* brow = &lds_vt[(t * 16 + m_) * 72 + 32 * s];
        const v16bf bfrag = pack2(brow + 8 * half, brow + 16 + 8 * half);
        oacc[t] = wmma_bf16(ap, bfrag, oacc[t]);
      }
    }
  }

  // Normalize and store to [B,N,D] bf16
#pragma unroll
  for (int i = 0; i < 8; ++i) {
    const int qrow = qbase + wave * 16 + i + 8 * half;
    const float invl = 1.0f / lrun[i];
#pragma unroll
    for (int t = 0; t < 4; ++t) {
      const int d = t * 16 + m_;
      ao[(size_t)(b * NTOK + qrow) * DIM + h * HD + d] = (bf16)(oacc[t][i] * invl);
    }
  }
}

// ---------------------------------------------------------------------------
// Kernel 3: OUT = AO @ Wo^T, fp32 output. grid (64, 16), block 128.
// AO tile is bf16 -> async staging; Wo needs fp32->bf16 convert -> ds path.
// ---------------------------------------------------------------------------
__global__ __launch_bounds__(128)
void out_proj_kernel(const bf16* __restrict__ ao,
                     const float* __restrict__ Wo,
                     float* __restrict__ out) {
  __shared__ __align__(16) bf16 lds_a[64 * 40];
  __shared__ __align__(16) bf16 lds_w[64 * 40];

  const int tid  = threadIdx.x;
  const int lane = tid & 31, wave = tid >> 5;
  const int m_   = lane & 15, half = lane >> 4;
  const int rowBase = blockIdx.x * 64;
  const int eBase   = blockIdx.y * 64;

  v8f acc[4] = {zero8(), zero8(), zero8(), zero8()};

  const int chunk = tid & 3;
  const int rhalf = tid >> 2;

  for (int kc = 0; kc < DIM; kc += 32) {
#pragma unroll
    for (int p = 0; p < 2; ++p) {
      const int row = p * 32 + rhalf;
      async_copy_b128(&lds_a[row * 40 + chunk * 8],
                      ao + (size_t)(rowBase + row) * DIM + kc + chunk * 8);
      const float4* ws = (const float4*)(Wo + (size_t)(eBase + row) * DIM + kc + chunk * 8);
      cvt_store8(&lds_w[row * 40 + chunk * 8], ws[0], ws[1]);
    }
    wait_async0();
    __syncthreads();

    const bf16* arow = &lds_a[(wave * 16 + m_) * 40];
    const v16bf afrag = pack2(arow + 8 * half, arow + 16 + 8 * half);
#pragma unroll
    for (int t = 0; t < 4; ++t) {
      const bf16* brow = &lds_w[(t * 16 + m_) * 40];
      const v16bf bfrag = pack2(brow + 8 * half, brow + 16 + 8 * half);
      acc[t] = wmma_bf16(afrag, bfrag, acc[t]);
    }
    __syncthreads();
  }

#pragma unroll
  for (int i = 0; i < 8; ++i) {
    const int gr = rowBase + wave * 16 + i + 8 * half;
#pragma unroll
    for (int t = 0; t < 4; ++t) {
      out[(size_t)gr * DIM + eBase + t * 16 + m_] = acc[t][i];
    }
  }
}

// ---------------------------------------------------------------------------
extern "C" void kernel_launch(void* const* d_in, const int* in_sizes, int n_in,
                              void* d_out, int out_size, void* d_ws, size_t ws_size,
                              hipStream_t stream) {
  (void)in_sizes; (void)n_in; (void)out_size; (void)ws_size;
  const float* x   = (const float*)d_in[0];
  const float* Wq  = (const float*)d_in[1];
  const float* Wk  = (const float*)d_in[2];
  const float* Wv  = (const float*)d_in[3];
  const float* Wo  = (const float*)d_in[4];
  const float* qnw = (const float*)d_in[5];
  const float* knw = (const float*)d_in[6];
  const int*   rp  = (const int*)d_in[7];

  const size_t perBuf = (size_t)BATCH * NHEAD * NTOK * HD;  // 4,194,304 elems
  bf16* qb = (bf16*)d_ws;
  bf16* kb = qb + perBuf;
  bf16* vb = kb + perBuf;
  bf16* ao = vb + perBuf;   // [B,N,D] == same element count

  dim3 blk(128);
  dim3 g1((BATCH * NTOK) / 64, NHEAD, 3);
  qkv_rope_norm_kernel<<<g1, blk, 0, stream>>>(x, Wq, Wk, Wv, qnw, knw, rp,
                                               qb, kb, vb);
  dim3 g2(NTOK / 64, NHEAD, BATCH);
  flash_attn_kernel<<<g2, blk, 0, stream>>>(qb, kb, vb, ao);
  dim3 g3((BATCH * NTOK) / 64, DIM / 64);
  out_proj_kernel<<<g3, blk, 0, stream>>>(ao, Wo, (float*)d_out);
}
